// BinarizeLinear_76716705841731
// MI455X (gfx1250) — matrix operand order repaired and checked by reading, not compiled, not run on hardware
//
#include <hip/hip_runtime.h>

// ---------------------------------------------------------------------------
// out = x @ sign(W)
//   x [32768,512] f32,  W [512,512] f32 ([k][n]),  out [32768,512] f32
//
// MI455X roofline: 17.2 GFLOP vs ~129 MB HBM -> ~5.5 us floor at 23.3 TB/s.
// f16 WMMA (f32 accum) is the cheapest matrix path that represents the
// binarized +-1 weights exactly; x's f16 conversion is the only rounding.
// Packed-binarized W (512 KB) is L2-resident (192 MB L2) for the whole GEMM.
//
// GEMM shape per wave: 32(M) x 64(N), K-pipelined (prefetch kb+1 fragments
// before issuing kb's 8 WMMAs) so WMMAs never wait on loadcnt==0.
// ---------------------------------------------------------------------------

typedef __attribute__((ext_vector_type(16))) _Float16 v16h;
typedef __attribute__((ext_vector_type(8)))  float    v8f;
typedef __attribute__((ext_vector_type(4)))  float    v4f;

#define IN_DIM   512
#define OUT_DIM  512
#define N_TOKENS 32768
#define KB_COUNT (IN_DIM / 32)    // 16 K-blocks of 32
#define NT_COUNT (OUT_DIM / 16)   // 32 N-tiles of 16

// ---------------------------------------------------------------------------
// Kernel 1: binarize W into WMMA B-operand register order (32x16 f16 frag,
// wave32): lane L -> column n0 + (L&15); half = L>>4; 16 f16 = K
// k0 + half*8 + 0..7, then k0 + 16 + half*8 + 0..7. Fragment = 32 lanes x 32 B = 1 KB,
// stored frag-major so the GEMM reads it as one fully-coalesced 1 KB block.
// ---------------------------------------------------------------------------
__global__ __launch_bounds__(256) void binarize_pack_kernel(
    const float* __restrict__ W, _Float16* __restrict__ Bp) {
  int t    = blockIdx.x * blockDim.x + threadIdx.x;  // 0 .. 16383
  int lane = t & 31;
  int frag = t >> 5;                 // 0 .. 511  (= nt*KB_COUNT + kb)
  int kb   = frag & (KB_COUNT - 1);
  int nt   = frag / KB_COUNT;
  int n    = nt * 16 + (lane & 15);
  int half = lane >> 4;
  int k0   = kb * 32 + half * 8;

  v16h v;
#pragma unroll
  for (int i = 0; i < 16; ++i) {
    float w = W[(size_t)(k0 + i + (i >= 8 ? 8 : 0)) * OUT_DIM + n];
    float s = (w > 0.0f) ? 1.0f : ((w < 0.0f) ? -1.0f : 0.0f);
    v[i] = (_Float16)s;
  }
  *(v16h*)(Bp + (size_t)frag * 512 + lane * 16) = v;
}

// A fragment (16x32 f16, ISA layout), p = &x[row][k0 + half*8]:
//   a[0..7]  = x[row][k0 + half*8 + 0..7]       (VGPRs 0-3)
//   a[8..15] = x[row][k0 + 16 + half*8 + 0..7]  (VGPRs 4-7)
__device__ __forceinline__ v16h load_a_frag(const float* __restrict__ p) {
  v4f a0 = *(const v4f*)(p);
  v4f a1 = *(const v4f*)(p + 4);
  v4f a2 = *(const v4f*)(p + 16);
  v4f a3 = *(const v4f*)(p + 20);
  v16h a;
#pragma unroll
  for (int i = 0; i < 4; ++i) {
    a[i]      = (_Float16)a0[i];
    a[4 + i]  = (_Float16)a1[i];
    a[8 + i]  = (_Float16)a2[i];
    a[12 + i] = (_Float16)a3[i];
  }
  return a;
}

__device__ __forceinline__ void wmma8(v8f acc[2][4], v16h a0, v16h a1,
                                      const v16h b[4]) {
#pragma unroll
  for (int t = 0; t < 4; ++t) {
    acc[0][t] = __builtin_amdgcn_wmma_f32_16x16x32_f16(
        false, a0, false, b[t], (short)0, acc[0][t], false, false);
    acc[1][t] = __builtin_amdgcn_wmma_f32_16x16x32_f16(
        false, a1, false, b[t], (short)0, acc[1][t], false, false);
  }
}

// ---------------------------------------------------------------------------
// Kernel 2: WMMA GEMM.
//   grid.x = N_TOKENS/32 = 1024 workgroups of 256 threads (8 wave32).
//   Workgroup -> 32 rows x full N=512.  Wave w -> N tiles [w*4, w*4+4),
//   both 16-row M tiles.  x streams from HBM exactly once; redundant A
//   fragment loads across the 8 waves hit WGP$.
// ---------------------------------------------------------------------------
__global__ __launch_bounds__(256) void bgemm_wmma_kernel(
    const float* __restrict__ X, const _Float16* __restrict__ Bp,
    float* __restrict__ Out) {
  const int lane = threadIdx.x & 31;
  const int wave = threadIdx.x >> 5;       // 0..7
  const int half = lane >> 4;
  const int m0   = blockIdx.x * 32;
  const int nt0  = wave * 4;

  const float* __restrict__ xr0 =
      X + (size_t)(m0 + (lane & 15)) * IN_DIM + half * 8;
  const float* __restrict__ xr1 = xr0 + (size_t)16 * IN_DIM;
  const _Float16* __restrict__ bb =
      Bp + (size_t)nt0 * KB_COUNT * 512 + lane * 16;

  v8f acc[2][4] = {};

  // Prologue: fragments for kb = 0.
  v16h a0 = load_a_frag(xr0);
  v16h a1 = load_a_frag(xr1);
  v16h b[4];
#pragma unroll
  for (int t = 0; t < 4; ++t)
    b[t] = *(const v16h*)(bb + (size_t)t * KB_COUNT * 512);

  // Pipelined main loop: prefetch kb+1 before the 8 WMMAs of kb.
#pragma unroll
  for (int kb = 0; kb < KB_COUNT - 1; ++kb) {
    v16h na0 = load_a_frag(xr0 + (kb + 1) * 32);
    v16h na1 = load_a_frag(xr1 + (kb + 1) * 32);
    v16h nb[4];
#pragma unroll
    for (int t = 0; t < 4; ++t)
      nb[t] = *(const v16h*)(bb + (size_t)(t * KB_COUNT + kb + 1) * 512);

    wmma8(acc, a0, a1, b);

    a0 = na0;
    a1 = na1;
#pragma unroll
    for (int t = 0; t < 4; ++t) b[t] = nb[t];
  }
  // Epilogue: last K-step.
  wmma8(acc, a0, a1, b);

  // C/D layout: lane L, VGPR r -> row m_base + half*8 + r, col n0 + (L&15).
  // Non-temporal: output written once, keep it out of the B-resident L2 set.
#pragma unroll
  for (int mt = 0; mt < 2; ++mt) {
#pragma unroll
    for (int t = 0; t < 4; ++t) {
      const int n = (nt0 + t) * 16 + (lane & 15);
#pragma unroll
      for (int r = 0; r < 8; ++r) {
        __builtin_nontemporal_store(
            acc[mt][t][r],
            Out + (size_t)(m0 + mt * 16 + half * 8 + r) * OUT_DIM + n);
      }
    }
  }
}

extern "C" void kernel_launch(void* const* d_in, const int* in_sizes, int n_in,
                              void* d_out, int out_size, void* d_ws, size_t ws_size,
                              hipStream_t stream) {
  const float* x = (const float*)d_in[0];   // [N_TOKENS, IN_DIM] f32
  const float* W = (const float*)d_in[1];   // [IN_DIM, OUT_DIM] f32
  float* out     = (float*)d_out;           // [N_TOKENS, OUT_DIM] f32
  _Float16* Bp   = (_Float16*)d_ws;         // 512 KB packed binarized weights

  binarize_pack_kernel<<<64, 256, 0, stream>>>(W, Bp);
  bgemm_wmma_kernel<<<N_TOKENS / 32, 256, 0, stream>>>(x, Bp, out);
}
